// rnn_decoder_11622181503698
// MI455X (gfx1250) — matrix operand order repaired and checked by reading, not compiled, not run on hardware
//
#include <hip/hip_runtime.h>
#include <hip/hip_bf16.h>
#include <cstdint>
#include <cstddef>

// ---------- problem constants ----------
#define TT 32
#define BB 32
#define HH 512
#define LL 2
#define SS 512
#define BH (BB * HH)        // 16384

typedef __attribute__((ext_vector_type(16))) _Float16 v16h;
typedef __attribute__((ext_vector_type(8)))  _Float16 v8h;
typedef __attribute__((ext_vector_type(8)))  float    v8f;

// ---------------------------------------------------------------------------
// WMMA fragment loaders (CDNA5 16x16x32 f16, wave32, contiguous k order in A and B alike)
// ---------------------------------------------------------------------------
// A: [16 x 32] row-major, lda in elements.
// lane 0-15  : row M=lane, K = 0..15
// lane 16-31 : row M=lane-16, K = 16..31
__device__ __forceinline__ v16h loadA(const _Float16* __restrict__ A, int lda) {
  const int lane = threadIdx.x & 31;
  const int m    = lane & 15;
  const int hi   = lane >> 4;
  const _Float16* r0 = A + m * lda;
  return *(const v16h*)(r0 + hi * 16);
}

// B: [32 x 16] (KxN) where B[k][n] = W[n*ldw + k] (W rows are the N columns,
// K contiguous in memory). lane 0-15: col N=lane, K=0..15; lane 16-31: K=16..31.
__device__ __forceinline__ v16h loadB(const _Float16* __restrict__ W, int ldw) {
  const int lane = threadIdx.x & 31;
  const int n    = lane & 15;
  const int hi   = lane >> 4;
  return *(const v16h*)(W + n * ldw + hi * 16);
}

__device__ __forceinline__ v8f wmma_step(v16h a, v16h b, v8f c) {
  return __builtin_amdgcn_wmma_f32_16x16x32_f16(false, a, false, b, (short)0, c,
                                                false, false);
}

__device__ __forceinline__ float sigm(float x) { return 1.f / (1.f + __expf(-x)); }

// ---------------------------------------------------------------------------
// f32 -> f16 conversion (weights), grid-stride
// ---------------------------------------------------------------------------
__global__ void k_f2h(const float* __restrict__ src, _Float16* __restrict__ dst, int n) {
  for (int i = blockIdx.x * blockDim.x + threadIdx.x; i < n; i += gridDim.x * blockDim.x)
    dst[i] = (_Float16)src[i];
}

// ---------------------------------------------------------------------------
// embedding gather -> f16  (one block per (t,b))
// ---------------------------------------------------------------------------
__global__ void k_embed(const int* __restrict__ inputs, const float* __restrict__ emb,
                        _Float16* __restrict__ embs16) {
  const int tb  = blockIdx.x;                 // T*B
  const int tok = inputs[tb];
  const float*    src = emb + (size_t)tok * HH;
  _Float16*       dst = embs16 + (size_t)tb * HH;
  for (int e = threadIdx.x; e < HH; e += blockDim.x) dst[e] = (_Float16)src[e];
}

// ---------------------------------------------------------------------------
// state init: h0/c0 -> f32 state + h f16 buffer 0
// ---------------------------------------------------------------------------
__global__ void k_init(const float* __restrict__ h0, const float* __restrict__ c0,
                       float* __restrict__ h_f32, float* __restrict__ c_state,
                       _Float16* __restrict__ h16) {
  int i = blockIdx.x * blockDim.x + threadIdx.x;   // L*B*H
  if (i < LL * BH) {
    h_f32[i]   = h0[i];
    c_state[i] = c0[i];
    int l = i / BH, r = i % BH;
    h16[(l * 2 + 0) * BH + r] = (_Float16)h0[i];   // double-buffer slot 0
  }
}

// ---------------------------------------------------------------------------
// Fused LSTM layer: gates = x@Wih^T + h@Whh^T + b, then cell update.
// grid.x = H/16 = 32 workgroups (each owns 16 columns of every gate),
// block = 256 (8 waves: wave w -> gate g=w&3, m-tile mt=w>>2).
// ---------------------------------------------------------------------------
__global__ void k_lstm(const _Float16* __restrict__ x16,    // [B,H]
                       const _Float16* __restrict__ hin16,  // [B,H]
                       const _Float16* __restrict__ Wih,    // [4H,H] f16
                       const _Float16* __restrict__ Whh,    // [4H,H] f16
                       const float* __restrict__ bih,       // [4H]
                       const float* __restrict__ bhh,       // [4H]
                       float* __restrict__ c_state,         // [B,H] in/out
                       float* __restrict__ h_f32,           // [B,H] out
                       _Float16* __restrict__ hout16) {     // [B,H] out
  __shared__ _Float16 sA[BB * HH];        // 32 KB staged A operand
  __shared__ float    gbuf[4][BB][16];    // gate tiles for cell exchange

  const int w  = threadIdx.x >> 5;
  const int g  = w & 3;
  const int mt = w >> 2;
  const int j0 = blockIdx.x * 16;
  const int r0 = g * HH + j0;

  v8f acc = {};
#pragma unroll
  for (int phase = 0; phase < 2; ++phase) {
    const _Float16* src = (phase == 0) ? x16 : hin16;
    const _Float16* Wm  = (phase == 0) ? Wih : Whh;
    __syncthreads();
    for (int i = threadIdx.x; i < BB * HH / 8; i += blockDim.x)
      ((float4*)sA)[i] = ((const float4*)src)[i];
    __syncthreads();
    const _Float16* Abase = sA + mt * 16 * HH;
    const _Float16* Wb    = Wm + (size_t)r0 * HH;
    // unroll capped at 2: full unroll (16x) spills B fragments to scratch
#pragma unroll 2
    for (int k0 = 0; k0 < HH; k0 += 32) {
      v16h a = loadA(Abase + k0, HH);
      v16h b = loadB(Wb + k0, HH);
      acc = wmma_step(a, b, acc);
    }
  }

  const int lane = threadIdx.x & 31;
  const int n  = lane & 15;
  const int mb = (lane >> 4) * 8;
#pragma unroll
  for (int v = 0; v < 8; ++v) gbuf[g][mt * 16 + mb + v][n] = acc[v];
  __syncthreads();

  // LSTM cell on the 32x16 column slab owned by this workgroup
  for (int e = threadIdx.x; e < BB * 16; e += blockDim.x) {
    const int row = e >> 4;
    const int col = e & 15;
    const int j   = j0 + col;
    float iv = gbuf[0][row][col] + bih[j]            + bhh[j];
    float fv = gbuf[1][row][col] + bih[HH + j]       + bhh[HH + j];
    float gv = gbuf[2][row][col] + bih[2 * HH + j]   + bhh[2 * HH + j];
    float ov = gbuf[3][row][col] + bih[3 * HH + j]   + bhh[3 * HH + j];
    float c_old = c_state[row * HH + j];
    float c2 = sigm(fv) * c_old + sigm(iv) * tanhf(gv);
    float h2 = sigm(ov) * tanhf(c2);
    c_state[row * HH + j] = c2;
    h_f32[row * HH + j]   = h2;
    hout16[row * HH + j]  = (_Float16)h2;
  }
}

// ---------------------------------------------------------------------------
// gamma = A16[32,512] @ Wi^T + bi   (Wi:[H,H] f16).  grid 8, block 256.
// wave w -> mt=w&1, nt=w>>1; 64 columns per workgroup.
// ---------------------------------------------------------------------------
__global__ void k_gamma(const _Float16* __restrict__ A16,
                        const _Float16* __restrict__ Wi16,
                        const float* __restrict__ bi,
                        float* __restrict__ gamma) {
  __shared__ _Float16 sA[BB * HH];
  for (int i = threadIdx.x; i < BB * HH / 8; i += blockDim.x)
    ((float4*)sA)[i] = ((const float4*)A16)[i];
  __syncthreads();

  const int w  = threadIdx.x >> 5;
  const int mt = w & 1;
  const int nt = w >> 1;
  const int j0 = blockIdx.x * 64 + nt * 16;

  v8f acc = {};
  const _Float16* Abase = sA + mt * 16 * HH;
  const _Float16* Wb    = Wi16 + (size_t)j0 * HH;
#pragma unroll 2
  for (int k0 = 0; k0 < HH; k0 += 32) {
    v16h a = loadA(Abase + k0, HH);
    v16h b = loadB(Wb + k0, HH);
    acc = wmma_step(a, b, acc);
  }

  const int lane = threadIdx.x & 31;
  const int n  = lane & 15;
  const int mb = (lane >> 4) * 8;
#pragma unroll
  for (int v = 0; v < 8; ++v) {
    int row = mt * 16 + mb + v, col = j0 + n;
    gamma[row * HH + col] = acc[v] + bi[col];
  }
}

// ---------------------------------------------------------------------------
// Per-batch context attention: scores = ctx[b] . gamma[b], weight (sigmoid or
// softmax), ct = sum_s w_s * ctx[b,s,:], writes concat16[b] = [ct | xpart].
// grid = B, block = 256.
// ---------------------------------------------------------------------------
__global__ void k_ctx(const float* __restrict__ ctx,     // [B,S,H]
                      const float* __restrict__ gamma,   // [B,H]
                      const _Float16* __restrict__ xpart16, // [B,H]
                      _Float16* __restrict__ concat16,   // [B,2H]
                      int softmax_mode) {
  const int b = blockIdx.x;
  __shared__ float sc[SS];
  __shared__ float sg[HH];
  __shared__ float red[256];

  for (int h = threadIdx.x; h < HH; h += blockDim.x) sg[h] = gamma[b * HH + h];
  __syncthreads();

  const float* cb = ctx + (size_t)b * SS * HH;
  for (int s = threadIdx.x; s < SS; s += blockDim.x) {
    const float4* row = (const float4*)(cb + (size_t)s * HH);
    const float4* g4  = (const float4*)sg;
    float acc = 0.f;
    for (int i = 0; i < HH / 4; ++i) {
      float4 r = row[i], g = g4[i];
      acc += r.x * g.x + r.y * g.y + r.z * g.z + r.w * g.w;
    }
    sc[s] = acc;
  }
  __syncthreads();

  if (softmax_mode) {
    float m = -3.4e38f;
    for (int s = threadIdx.x; s < SS; s += blockDim.x) m = fmaxf(m, sc[s]);
    red[threadIdx.x] = m;
    __syncthreads();
    for (int off = 128; off > 0; off >>= 1) {
      if (threadIdx.x < off) red[threadIdx.x] = fmaxf(red[threadIdx.x], red[threadIdx.x + off]);
      __syncthreads();
    }
    float mx = red[0];
    __syncthreads();
    float sum = 0.f;
    for (int s = threadIdx.x; s < SS; s += blockDim.x) {
      float e = __expf(sc[s] - mx);
      sc[s] = e;
      sum += e;
    }
    red[threadIdx.x] = sum;
    __syncthreads();
    for (int off = 128; off > 0; off >>= 1) {
      if (threadIdx.x < off) red[threadIdx.x] += red[threadIdx.x + off];
      __syncthreads();
    }
    float inv = 1.f / red[0];
    __syncthreads();
    for (int s = threadIdx.x; s < SS; s += blockDim.x) sc[s] *= inv;
  } else {
    for (int s = threadIdx.x; s < SS; s += blockDim.x) sc[s] = sigm(sc[s]);
  }
  __syncthreads();

  for (int h = threadIdx.x; h < HH; h += blockDim.x) {
    float acc = 0.f;
    for (int s = 0; s < SS; ++s) acc += sc[s] * cb[(size_t)s * HH + h];
    concat16[(size_t)b * 2 * HH + h] = (_Float16)acc;
  }
  for (int h = threadIdx.x; h < HH; h += blockDim.x)
    concat16[(size_t)b * 2 * HH + HH + h] = xpart16[b * HH + h];
}

// ---------------------------------------------------------------------------
// out = tanh(C16[32,1024] @ Wo^T + bo) (+resid). grid 8, block 256, K=1024.
// ---------------------------------------------------------------------------
__global__ void k_outp(const _Float16* __restrict__ C16,   // [32,1024]
                       const _Float16* __restrict__ Wo16,  // [512,1024]
                       const float* __restrict__ bo,
                       const float* __restrict__ resid,    // nullable
                       float* __restrict__ dstf,           // [32,512]
                       _Float16* __restrict__ dsth) {      // [32,512]
  __shared__ _Float16 sA[BB * HH];   // one 512-wide K chunk at a time
  const int w  = threadIdx.x >> 5;
  const int mt = w & 1;
  const int nt = w >> 1;
  const int j0 = blockIdx.x * 64 + nt * 16;

  v8f acc = {};
#pragma unroll
  for (int kk = 0; kk < 2; ++kk) {
    __syncthreads();
    for (int i = threadIdx.x; i < BB * HH / 8; i += blockDim.x) {
      int r = i >> 6, cq = i & 63;   // 64 float4 per 512-wide row
      ((float4*)sA)[i] = ((const float4*)(C16 + (size_t)r * 2 * HH + kk * HH))[cq];
    }
    __syncthreads();
    const _Float16* Abase = sA + mt * 16 * HH;
    const _Float16* Wb    = Wo16 + (size_t)j0 * 2 * HH + kk * HH;
#pragma unroll 2
    for (int k0 = 0; k0 < HH; k0 += 32) {
      v16h a = loadA(Abase + k0, HH);
      v16h b = loadB(Wb + k0, 2 * HH);
      acc = wmma_step(a, b, acc);
    }
  }

  const int lane = threadIdx.x & 31;
  const int n  = lane & 15;
  const int mb = (lane >> 4) * 8;
#pragma unroll
  for (int v = 0; v < 8; ++v) {
    int row = mt * 16 + mb + v, col = j0 + n;
    float val = tanhf(acc[v] + bo[col]);
    if (resid) val += resid[row * HH + col];
    dstf[row * HH + col] = val;
    dsth[row * HH + col] = (_Float16)val;
  }
}

// ---------------------------------------------------------------------------
// final state copy into d_out tail
// ---------------------------------------------------------------------------
__global__ void k_fin(const float* __restrict__ h_f32, const float* __restrict__ c_state,
                      float* __restrict__ tail) {
  int i = blockIdx.x * blockDim.x + threadIdx.x;
  if (i < LL * BH) {
    tail[i]           = h_f32[i];
    tail[LL * BH + i] = c_state[i];
  }
}

// ---------------------------------------------------------------------------
extern "C" void kernel_launch(void* const* d_in, const int* in_sizes, int n_in,
                              void* d_out, int out_size, void* d_ws, size_t ws_size,
                              hipStream_t stream) {
  (void)in_sizes; (void)n_in; (void)out_size; (void)ws_size;
  const int*   inputs   = (const int*)d_in[0];
  const float* h0       = (const float*)d_in[1];
  const float* c0       = (const float*)d_in[2];
  const float* contexts = (const float*)d_in[3];
  const float* ctx_conv = (const float*)d_in[4];
  const float* emb      = (const float*)d_in[5];
  const float* W_ih     = (const float*)d_in[6];
  const float* W_hh     = (const float*)d_in[7];
  const float* b_ih     = (const float*)d_in[8];
  const float* b_hh     = (const float*)d_in[9];
  const float* Wi_att   = (const float*)d_in[10];
  const float* bi_att   = (const float*)d_in[11];
  const float* Wo_att   = (const float*)d_in[12];
  const float* bo_att   = (const float*)d_in[13];
  const float* Wi_conv  = (const float*)d_in[14];
  const float* bi_conv  = (const float*)d_in[15];
  const float* Wo_conv  = (const float*)d_in[16];
  const float* bo_conv  = (const float*)d_in[17];
  float* dout = (float*)d_out;

  // ---- workspace carve-up (bytes, 256B aligned) ----
  char* ws = (char*)d_ws;
  size_t off = 0;
  auto carve = [&](size_t bytes) { char* p = ws + off; off = (off + bytes + 255) & ~(size_t)255; return p; };
  _Float16* Wih16    = (_Float16*)carve((size_t)LL * 4 * HH * HH * 2);
  _Float16* Whh16    = (_Float16*)carve((size_t)LL * 4 * HH * HH * 2);
  _Float16* Wiatt16  = (_Float16*)carve((size_t)HH * HH * 2);
  _Float16* Woatt16  = (_Float16*)carve((size_t)HH * 2 * HH * 2);
  _Float16* Wiconv16 = (_Float16*)carve((size_t)HH * HH * 2);
  _Float16* Woconv16 = (_Float16*)carve((size_t)HH * 2 * HH * 2);
  _Float16* embs16   = (_Float16*)carve((size_t)TT * BH * 2);
  _Float16* h16      = (_Float16*)carve((size_t)LL * 2 * BH * 2);  // double-buffered
  float*    h_f32    = (float*)carve((size_t)LL * BH * 4);
  float*    c_state  = (float*)carve((size_t)LL * BH * 4);
  float*    gamma    = (float*)carve((size_t)BH * 4);
  _Float16* concat16 = (_Float16*)carve((size_t)BB * 2 * HH * 2);
  float*    outc_f32 = (float*)carve((size_t)BH * 4);
  _Float16* outc16   = (_Float16*)carve((size_t)BH * 2);
  _Float16* scr16    = (_Float16*)carve((size_t)BH * 2);

  // ---- one-time per-launch preprocessing ----
  k_f2h<<<1024, 256, 0, stream>>>(W_ih,    Wih16,    LL * 4 * HH * HH);
  k_f2h<<<1024, 256, 0, stream>>>(W_hh,    Whh16,    LL * 4 * HH * HH);
  k_f2h<<<512,  256, 0, stream>>>(Wi_att,  Wiatt16,  HH * HH);
  k_f2h<<<512,  256, 0, stream>>>(Wo_att,  Woatt16,  HH * 2 * HH);
  k_f2h<<<512,  256, 0, stream>>>(Wi_conv, Wiconv16, HH * HH);
  k_f2h<<<512,  256, 0, stream>>>(Wo_conv, Woconv16, HH * 2 * HH);
  k_embed<<<TT * BB, 256, 0, stream>>>(inputs, emb, embs16);
  k_init<<<(LL * BH + 255) / 256, 256, 0, stream>>>(h0, c0, h_f32, c_state, h16);

  // ---- timestep loop ----
  int cur = 0;
  for (int t = 0; t < TT; ++t) {
    int nxt = cur ^ 1;
    // layer 0
    k_lstm<<<HH / 16, 256, 0, stream>>>(
        embs16 + (size_t)t * BH, h16 + (size_t)(0 * 2 + cur) * BH,
        Wih16, Whh16, b_ih, b_hh,
        c_state, h_f32, h16 + (size_t)(0 * 2 + nxt) * BH);
    // layer 1
    k_lstm<<<HH / 16, 256, 0, stream>>>(
        h16 + (size_t)(0 * 2 + nxt) * BH, h16 + (size_t)(1 * 2 + cur) * BH,
        Wih16 + (size_t)4 * HH * HH, Whh16 + (size_t)4 * HH * HH,
        b_ih + 4 * HH, b_hh + 4 * HH,
        c_state + BH, h_f32 + BH, h16 + (size_t)(1 * 2 + nxt) * BH);

    const _Float16* x16 = h16 + (size_t)(1 * 2 + nxt) * BH;

    // conv attention (sigmoid weights)
    k_gamma<<<8, 256, 0, stream>>>(x16, Wiconv16, bi_conv, gamma);
    k_ctx<<<BB, 256, 0, stream>>>(ctx_conv, gamma, x16, concat16, 0);
    k_outp<<<8, 256, 0, stream>>>(concat16, Woconv16, bo_conv, nullptr, outc_f32, outc16);

    // softmax attention, query = out_conv, residual add
    k_gamma<<<8, 256, 0, stream>>>(outc16, Wiatt16, bi_att, gamma);
    k_ctx<<<BB, 256, 0, stream>>>(contexts, gamma, outc16, concat16, 1);
    k_outp<<<8, 256, 0, stream>>>(concat16, Woatt16, bo_att, outc_f32,
                                  dout + (size_t)t * BH, scr16);
    cur = nxt;
  }

  k_fin<<<(LL * BH + 255) / 256, 256, 0, stream>>>(h_f32, c_state, dout + (size_t)TT * BH);
}
